// MLP_Diag_14285061227128
// MI455X (gfx1250) — matrix operand order repaired and checked by reading, not compiled, not run on hardware
//
#include <hip/hip_runtime.h>
#include <hip/hip_bf16.h>

typedef __attribute__((ext_vector_type(16))) __bf16 v16bf;
typedef __attribute__((ext_vector_type(8)))  __bf16 v8bf;
typedef __attribute__((ext_vector_type(8)))  float  v8f;

#define N_ROWS 8192
#define DIM    512
#define KEEP   31   // top_k keeps K+1 = 31 entries per row

// ---------------------------------------------------------------------------
// Pass A: emb = normalize(relu(x*W1)*W2), stored as bf16 row-major E[N][D]
// ---------------------------------------------------------------------------
__global__ void __launch_bounds__(256)
emb_kernel(const float* __restrict__ feat, const float* __restrict__ W1,
           const float* __restrict__ W2, __hip_bfloat16* __restrict__ E) {
    __shared__ float red[256];
    __shared__ float invn;
    const int row = blockIdx.x;
    const int t = threadIdx.x;
    const int c0 = t, c1 = t + 256;            // D=512 -> 2 cols per thread
    float h0 = fmaxf(feat[row * DIM + c0] * W1[c0], 0.0f) * W2[c0];
    float h1 = fmaxf(feat[row * DIM + c1] * W1[c1], 0.0f) * W2[c1];
    red[t] = h0 * h0 + h1 * h1;
    __syncthreads();
    for (int s = 128; s > 0; s >>= 1) {
        if (t < s) red[t] += red[t + s];
        __syncthreads();
    }
    if (t == 0) invn = 1.0f / fmaxf(sqrtf(red[0]), 1e-12f);
    __syncthreads();
    const float r = invn;
    E[row * DIM + c0] = __float2bfloat16(h0 * r);
    E[row * DIM + c1] = __float2bfloat16(h1 * r);
}

// ---------------------------------------------------------------------------
// Pass B: LDS-tiled transpose  E[N][D] (bf16) -> ET[D][N] (bf16)
// (ET is unused by the WMMA GEMM, which loads B-operands from row-major E)
// ---------------------------------------------------------------------------
__global__ void __launch_bounds__(256)
transpose_kernel(const unsigned short* __restrict__ E,
                 unsigned short* __restrict__ ET) {
    __shared__ unsigned short tile[32][33];    // +1 pad: no bank conflicts
    const int x  = threadIdx.x & 31;
    const int y0 = threadIdx.x >> 5;           // 0..7
    const int c0 = blockIdx.x * 32;            // D direction
    const int r0 = blockIdx.y * 32;            // N direction
#pragma unroll
    for (int i = 0; i < 32; i += 8)
        tile[y0 + i][x] = E[(size_t)(r0 + y0 + i) * DIM + c0 + x];
    __syncthreads();
#pragma unroll
    for (int i = 0; i < 32; i += 8)
        ET[(size_t)(c0 + y0 + i) * N_ROWS + r0 + x] = tile[x][y0 + i];
}

// ---------------------------------------------------------------------------
// Pass C: sim = E @ E^T via v_wmma_f32_16x16x32_bf16.
// Wave -> 32x64 C slab (2 A-frags x 4 B-frags, 8 f32 accumulators).
// Block (8 waves) -> 32x512. Grid (256, 16) covers 8192x8192.
// ---------------------------------------------------------------------------
__global__ void __launch_bounds__(256)
simgemm_kernel(const __hip_bfloat16* __restrict__ Eh,
               const __hip_bfloat16* __restrict__ ETh,
               float* __restrict__ out) {
    const __bf16* E  = (const __bf16*)Eh;
    const __bf16* ET = (const __bf16*)ETh;
    const int lane = threadIdx.x & 31;
    const int wave = threadIdx.x >> 5;
    const int i0 = blockIdx.x * 32;                    // row tiles i0, i0+16
    const int j0 = blockIdx.y * 512 + wave * 64;       // four col tiles

    // A layout: lanes 0-15 -> M=lane, K in {0..7,16..23}; lanes 16-31 -> M=lane-16, K+8
    const int arow   = i0 + (lane & 15);
    const int kshift = (lane >> 4) << 3;               // 0 or 8

    v8f acc[2][4];
#pragma unroll
    for (int a = 0; a < 2; ++a)
#pragma unroll
        for (int b = 0; b < 4; ++b) acc[a][b] = v8f{};

    for (int k0 = 0; k0 < DIM; k0 += 32) {
        // A fragments: two contiguous 16B chunks per lane from row-major E
        v16bf afrag[2];
#pragma unroll
        for (int a = 0; a < 2; ++a) {
            const __bf16* base = E + (size_t)(arow + a * 16) * DIM + k0 + kshift;
            v8bf lo = *(const v8bf*)(base);            // K = kbase+0..7
            v8bf hi = *(const v8bf*)(base + 16);       // K = kbase+16..23
            afrag[a] = __builtin_shufflevector(lo, hi, 0, 1, 2, 3, 4, 5, 6, 7,
                                               8, 9, 10, 11, 12, 13, 14, 15);
        }
        // B layout: N = lane&15, K same order as A (kshift + {0..7,16..23}) from row-major E
        const __bf16* brow = E + (size_t)(j0 + (lane & 15)) * DIM + k0 + kshift;
        if (k0 + 32 < DIM)                              // global_prefetch_b8
            __builtin_prefetch(brow + 32, 0, 0);
#pragma unroll
        for (int t = 0; t < 4; ++t) {
            v16bf bfrag = __builtin_shufflevector(*(const v8bf*)(brow + (size_t)t * 16 * DIM),
                                                  *(const v8bf*)(brow + (size_t)t * 16 * DIM + 16),
                                                  0, 1, 2, 3, 4, 5, 6, 7,
                                                  8, 9, 10, 11, 12, 13, 14, 15);
            acc[0][t] = __builtin_amdgcn_wmma_f32_16x16x32_bf16(
                false, afrag[0], false, bfrag, (short)0, acc[0][t], false, false);
            acc[1][t] = __builtin_amdgcn_wmma_f32_16x16x32_bf16(
                false, afrag[1], false, bfrag, (short)0, acc[1][t], false, false);
        }
    }

    // C layout: VGPR r -> M = r + 8*(lane>=16), N = lane&15
    const int n     = lane & 15;
    const int mbase = (lane >> 4) << 3;
#pragma unroll
    for (int a = 0; a < 2; ++a)
#pragma unroll
        for (int t = 0; t < 4; ++t)
#pragma unroll
            for (int r = 0; r < 8; ++r)
                out[(size_t)(i0 + a * 16 + mbase + r) * N_ROWS + (j0 + t * 16 + n)]
                    = acc[a][t][r];
}

// ---------------------------------------------------------------------------
// Pass D: per-row top-31 threshold via bisection on order-preserving float
// keys (row staged once in LDS), then masked relu written back in place.
// ---------------------------------------------------------------------------
__device__ __forceinline__ unsigned fkey(float f) {
    unsigned u = __float_as_uint(f);
    return (u & 0x80000000u) ? ~u : (u | 0x80000000u);   // monotone total order
}

__global__ void __launch_bounds__(256)
topk_kernel(float* __restrict__ out) {
    __shared__ float row[N_ROWS];                       // 32 KB of 320 KB LDS
    __shared__ unsigned cnt;
    const int t = threadIdx.x;
    float* rp = out + (size_t)blockIdx.x * N_ROWS;
#pragma unroll
    for (int i = 0; i < N_ROWS / 256; ++i)
        row[t + i * 256] = rp[t + i * 256];
    __syncthreads();

    // find max key thr with |{v : key(v) >= thr}| >= KEEP  => key of 31st largest
    unsigned lo = 0u, hi = 0xFFFFFFFFu;
    while (lo < hi) {
        unsigned long long span = (unsigned long long)(hi - lo) + 1ull;
        unsigned mid = lo + (unsigned)(span >> 1);
        if (t == 0) cnt = 0u;
        __syncthreads();
        unsigned local = 0;
#pragma unroll
        for (int i = 0; i < N_ROWS / 256; ++i)
            local += (fkey(row[t + i * 256]) >= mid) ? 1u : 0u;
        atomicAdd(&cnt, local);
        __syncthreads();
        unsigned c = cnt;
        __syncthreads();
        if (c >= KEEP) lo = mid; else hi = mid - 1;
    }

#pragma unroll
    for (int i = 0; i < N_ROWS / 256; ++i) {
        float v = row[t + i * 256];
        rp[t + i * 256] = (fkey(v) >= lo) ? fmaxf(v, 0.0f) : 0.0f;
    }
}

// ---------------------------------------------------------------------------
extern "C" void kernel_launch(void* const* d_in, const int* in_sizes, int n_in,
                              void* d_out, int out_size, void* d_ws, size_t ws_size,
                              hipStream_t stream) {
    (void)in_sizes; (void)n_in; (void)out_size; (void)ws_size;
    const float* feat = (const float*)d_in[0];
    const float* W1   = (const float*)d_in[1];
    const float* W2   = (const float*)d_in[2];
    float* out        = (float*)d_out;

    __hip_bfloat16* E  = (__hip_bfloat16*)d_ws;                          // 8 MB
    __hip_bfloat16* ET = (__hip_bfloat16*)((char*)d_ws +
                         (size_t)N_ROWS * DIM * sizeof(__hip_bfloat16)); // 8 MB

    emb_kernel<<<N_ROWS, 256, 0, stream>>>(feat, W1, W2, E);
    transpose_kernel<<<dim3(DIM / 32, N_ROWS / 32), 256, 0, stream>>>(
        (const unsigned short*)E, (unsigned short*)ET);
    simgemm_kernel<<<dim3(N_ROWS / 32, N_ROWS / 512), 256, 0, stream>>>(E, ET, out);
    topk_kernel<<<N_ROWS, 256, 0, stream>>>(out);
}
